// Attention_56770877718621
// MI455X (gfx1250) — matrix operand order repaired and checked by reading, not compiled, not run on hardware
//
#include <hip/hip_runtime.h>
#include <stdint.h>

typedef __attribute__((ext_vector_type(16))) _Float16 v16h;
typedef __attribute__((ext_vector_type(8)))  _Float16 v8h;
typedef __attribute__((ext_vector_type(4)))  _Float16 v4h;
typedef __attribute__((ext_vector_type(8)))  float    v8f;
typedef __attribute__((ext_vector_type(4)))  unsigned int u32x4;
typedef __attribute__((ext_vector_type(8)))  int      i32x8;
typedef __attribute__((ext_vector_type(4)))  int      i32x4;

union Frag16 { v16h f; v8h h[2]; };

__device__ __forceinline__ v8f zero8() {
  v8f z;
#pragma unroll
  for (int i = 0; i < 8; ++i) z[i] = 0.0f;
  return z;
}

#define GBM 128
#define GBN 128
#define GBK 32
#define LDSS 40  // halves per padded LDS row (64B data + 16B pad = 80B)

// ---------------------------------------------------------------------------
// TDM: DMA one [nrows x 32] f16 tile (row stride = kelems) from global into
// LDS at byte offset lds_off, inserting 16B of padding after every 64B row
// (pad_interval code 3 = 16 DWORDs, pad_amount code 3 = 4 DWORDs) so rows
// land exactly LDSS=40 halves apart. Tracked by TENSORcnt.
// ---------------------------------------------------------------------------
__device__ __forceinline__ void tdm_load_tile(const _Float16* gp, unsigned lds_off,
                                              unsigned nrows, unsigned kelems) {
  const unsigned long long ga = (unsigned long long)(uintptr_t)gp;
  u32x4 g0;
  g0[0] = 1u;                                                  // count=1, user D#
  g0[1] = lds_off;                                             // lds_addr (bytes)
  g0[2] = (unsigned)ga;                                        // global_addr[31:0]
  g0[3] = (unsigned)((ga >> 32) & 0x01FFFFFFu) | 0x80000000u;  // addr hi | type=2
  i32x8 g1;
  g1[0] = (int)((1u << 16) | (1u << 20) | (3u << 22) | (3u << 25)); // 2B elems, pad on
  g1[1] = (int)((kelems & 0xFFFFu) << 16);                          // tensor_dim0 lo
  g1[2] = (int)(((kelems >> 16) & 0xFFFFu) | ((nrows & 0xFFFFu) << 16)); // dim0 hi|dim1 lo
  g1[3] = (int)(((nrows >> 16) & 0xFFFFu) | ((unsigned)GBK << 16));      // dim1 hi|tile_dim0
  g1[4] = (int)nrows;                                               // tile_dim1 (tile_dim2=0)
  g1[5] = (int)kelems;                                              // tensor_dim0_stride lo
  g1[6] = 0;
  g1[7] = 0;
  i32x4 z4; z4[0] = 0; z4[1] = 0; z4[2] = 0; z4[3] = 0;
  i32x8 z8;
#pragma unroll
  for (int i = 0; i < 8; ++i) z8[i] = 0;
  // 6-arg form (clang-23 / therock headers): (g0, g1, g2, g3, g4, cpol)
  __builtin_amdgcn_tensor_load_to_lds(g0, g1, z4, z4, z8, 0);
}

// ---------------------------------------------------------------------------
// GEMM: C[M,N] = A[M,K] @ B[N,K]^T, A/B f16, fp32 WMMA accumulation.
// Block: 256 threads = 8 waves (4x2), tile 128x128x32, TDM double buffering.
// ---------------------------------------------------------------------------
__global__ __launch_bounds__(256)
void gemm_wmma_f16(const _Float16* __restrict__ A, const _Float16* __restrict__ B,
                   float* __restrict__ C, int M, int N, int K) {
  __shared__ _Float16 lds_pool[(2 * GBM + 2 * GBN) * LDSS];

  const int tid  = threadIdx.x;
  const int lane = tid & 31;
  const int wave = tid >> 5;
  const int wm = wave >> 1;   // 0..3
  const int wn = wave & 1;    // 0..1
  const int bm = blockIdx.y * GBM;
  const int bn = blockIdx.x * GBN;
  const int lr  = lane & 15;
  const int lhi = lane >> 4;
  const bool issuer = (wave == 0);

  v8f acc[2][4];
#pragma unroll
  for (int i = 0; i < 2; ++i)
#pragma unroll
    for (int j = 0; j < 4; ++j) acc[i][j] = zero8();

  if (issuer) {
    tdm_load_tile(A + (size_t)bm * K, 0u, GBM, (unsigned)K);
    tdm_load_tile(B + (size_t)bn * K, (unsigned)(2 * GBM * LDSS * 2), GBN, (unsigned)K);
  }

  int buf = 0;
  for (int kb = 0; kb < K; kb += GBK, buf ^= 1) {
    __builtin_amdgcn_s_wait_tensorcnt(0);  // no-op for non-issuing waves
    // Opaque writer: TDM filled lds_pool behind the compiler's back. Make the
    // LDS base escape into an asm with a memory clobber so the fragment loads
    // below cannot be undef-folded or hoisted above this point.
    {
      const _Float16* lp = &lds_pool[0];
      asm volatile("" : "+v"(lp)::"memory");
    }
    __syncthreads();                       // current buffer visible to all waves

    if (issuer && (kb + GBK < K)) {        // prefetch next tile into other buffer
      tdm_load_tile(A + (size_t)bm * K + kb + GBK,
                    (unsigned)((buf ^ 1) * GBM * LDSS * 2), GBM, (unsigned)K);
      tdm_load_tile(B + (size_t)bn * K + kb + GBK,
                    (unsigned)((2 * GBM + (buf ^ 1) * GBN) * LDSS * 2), GBN, (unsigned)K);
    }

    const unsigned abase = (unsigned)buf * (unsigned)(GBM * LDSS);
    const unsigned bbase = (unsigned)(2 * GBM * LDSS) + (unsigned)buf * (unsigned)(GBN * LDSS);

    // A fragments: lane lr = row m; chunk0 K=lhi*8.., chunk1 K=16+lhi*8..
    Frag16 af[2];
#pragma unroll
    for (int mi = 0; mi < 2; ++mi) {
      const unsigned ra = abase + (unsigned)((wm * 32 + mi * 16 + lr) * LDSS);
      af[mi].h[0] = *(const v8h*)(&lds_pool[ra + lhi * 8]);
      af[mi].h[1] = *(const v8h*)(&lds_pool[ra + 16 + lhi * 8]);
    }
    // B fragments: lane lr = col n; chunk0 K=lhi*8.., chunk1 K=16+lhi*8..
    Frag16 bf[4];
#pragma unroll
    for (int ni = 0; ni < 4; ++ni) {
      const unsigned rb = bbase + (unsigned)((wn * 64 + ni * 16 + lr) * LDSS);
      bf[ni].h[0] = *(const v8h*)(&lds_pool[rb + lhi * 8]);
      bf[ni].h[1] = *(const v8h*)(&lds_pool[rb + 16 + lhi * 8]);
    }
#pragma unroll
    for (int mi = 0; mi < 2; ++mi)
#pragma unroll
      for (int ni = 0; ni < 4; ++ni)
        acc[mi][ni] = __builtin_amdgcn_wmma_f32_16x16x32_f16(
            false, af[mi].f, false, bf[ni].f, (short)0, acc[mi][ni], false, false);

    __syncthreads();  // all reads of this buffer done before TDM may overwrite it
  }

  // C layout: lane 0-15 reg r -> (M=r, N=lane); lane 16-31 -> (M=8+r, N=lane-16)
#pragma unroll
  for (int mi = 0; mi < 2; ++mi)
#pragma unroll
    for (int ni = 0; ni < 4; ++ni) {
      const int n = bn + wn * 64 + ni * 16 + lr;
#pragma unroll
      for (int r = 0; r < 8; ++r) {
        const int m = bm + wm * 32 + mi * 16 + lhi * 8 + r;
        C[(size_t)m * N + n] = acc[mi][ni][r];
      }
    }
}

// ---------------------------------------------------------------------------
// fp32 -> f16 conversion, 4 elements/thread.
// ---------------------------------------------------------------------------
__global__ void cvt_f16_kernel(const float* __restrict__ src, _Float16* __restrict__ dst) {
  const size_t i = (size_t)(blockIdx.x * blockDim.x + threadIdx.x) * 4;
  const float4 v = *(const float4*)(src + i);
  v4h h = { (_Float16)v.x, (_Float16)v.y, (_Float16)v.z, (_Float16)v.w };
  *(v4h*)(dst + i) = h;
}

// ---------------------------------------------------------------------------
// RoPE on fused QKV output (row stride 6144). Q folds in 1/sqrt(128).
// ---------------------------------------------------------------------------
__global__ void rope_q_kernel(const float* __restrict__ QKV, const float* __restrict__ fc,
                              const float* __restrict__ fs, _Float16* __restrict__ Y) {
  const int idx = blockIdx.x * blockDim.x + threadIdx.x;  // S*32*64 pairs
  const int s = idx >> 11;
  const int rem = idx & 2047;
  const int h = rem >> 6;
  const int p = rem & 63;
  const float c = fc[s * 64 + p], sn = fs[s * 64 + p];
  const size_t bi = (size_t)s * 6144 + h * 128 + 2 * p;
  const size_t bo = (size_t)s * 4096 + h * 128 + 2 * p;
  const float xr = QKV[bi], xi = QKV[bi + 1];
  const float sc = 0.08838834764831845f;  // 1/sqrt(128)
  Y[bo]     = (_Float16)((xr * c - xi * sn) * sc);
  Y[bo + 1] = (_Float16)((xr * sn + xi * c) * sc);
}

__global__ void rope_k_kernel(const float* __restrict__ QKV, const float* __restrict__ fc,
                              const float* __restrict__ fs, _Float16* __restrict__ Y) {
  const int idx = blockIdx.x * blockDim.x + threadIdx.x;  // S*8*64 pairs
  const int s = idx >> 9;
  const int rem = idx & 511;
  const int h = rem >> 6;
  const int p = rem & 63;
  const float c = fc[s * 64 + p], sn = fs[s * 64 + p];
  const size_t bi = (size_t)s * 6144 + 4096 + h * 128 + 2 * p;
  const size_t bo = (size_t)s * 1024 + h * 128 + 2 * p;
  const float xr = QKV[bi], xi = QKV[bi + 1];
  Y[bo]     = (_Float16)(xr * c - xi * sn);
  Y[bo + 1] = (_Float16)(xr * sn + xi * c);
}

// V columns of fused QKV -> VT [1024, 2048] f16 (dim-major for PV B-frags)
__global__ void transpose_v_kernel(const float* __restrict__ QKV, _Float16* __restrict__ VT) {
  const int idx = blockIdx.x * blockDim.x + threadIdx.x;  // 1024*2048
  const int d = idx >> 11;
  const int s = idx & 2047;
  VT[idx] = (_Float16)QKV[(size_t)s * 6144 + 5120 + d];
}

// ---------------------------------------------------------------------------
// Flash attention. Block = 128 threads = 4 waves; each wave owns 16 q rows of
// one head. 32 keys/iter: 8 score WMMAs (2 tiles x 4 k-steps), online softmax (shfl_xor reductions),
// P transposed through per-wave LDS, 8 PV WMMAs. Output written in f16.
// ---------------------------------------------------------------------------
#define NEG_BIG (-1.0e30f)

__global__ __launch_bounds__(128)
void flash_attn_kernel(const _Float16* __restrict__ Qh,  // [S, 4096] (scaled)
                       const _Float16* __restrict__ Kh,  // [S, 1024]
                       const _Float16* __restrict__ VT,  // [1024, S]
                       _Float16* __restrict__ Out) {     // [S, 4096] f16
  __shared__ _Float16 Pl[4][16 * LDSS];

  const int tid  = threadIdx.x;
  const int lane = tid & 31;
  const int wave = tid >> 5;
  const int lr   = lane & 15;
  const int lhi  = lane >> 4;

  const int head = blockIdx.y;
  const int kvh  = head >> 2;            // N_REP = 4
  const int q0   = blockIdx.x * 64 + wave * 16;

  Frag16 qf[4];
  {
    const _Float16* qp = Qh + (size_t)(q0 + lr) * 4096 + head * 128;
#pragma unroll
    for (int f = 0; f < 4; ++f) {
      qf[f].h[0] = *(const v8h*)(qp + f * 32 + lhi * 16);
      qf[f].h[1] = *(const v8h*)(qp + f * 32 + lhi * 16 + 8);
    }
  }

  v8f acc[8];
#pragma unroll
  for (int t = 0; t < 8; ++t) acc[t] = zero8();
  float mrow[8], lsum[8];
#pragma unroll
  for (int r = 0; r < 8; ++r) { mrow[r] = NEG_BIG; lsum[r] = 0.0f; }

  const int kv_end = q0 + 16;
  for (int jb = 0; jb < kv_end; jb += 32) {
    v8f c0 = zero8(), c1 = zero8();
#pragma unroll
    for (int f = 0; f < 4; ++f) {
      Frag16 kf0, kf1;
      const _Float16* kp0 = Kh + (size_t)(jb + lr) * 1024 + kvh * 128 + f * 32 + lhi * 16;
      kf0.f = *(const v16h*)kp0;
      kf1.f = *(const v16h*)(kp0 + 16 * 1024);
      c0 = __builtin_amdgcn_wmma_f32_16x16x32_f16(false, qf[f].f, false, kf0.f,
                                                  (short)0, c0, false, false);
      c1 = __builtin_amdgcn_wmma_f32_16x16x32_f16(false, qf[f].f, false, kf1.f,
                                                  (short)0, c1, false, false);
    }

    if (jb + 31 > q0) {  // causal mask, diagonal tiles only
#pragma unroll
      for (int r = 0; r < 8; ++r) {
        const int q = q0 + lhi * 8 + r;
        if (jb + lr > q)      c0[r] = NEG_BIG;
        if (jb + 16 + lr > q) c1[r] = NEG_BIG;
      }
    }

#pragma unroll
    for (int r = 0; r < 8; ++r) {
      float mt = fmaxf(c0[r], c1[r]);
      mt = fmaxf(mt, __shfl_xor(mt, 1));
      mt = fmaxf(mt, __shfl_xor(mt, 2));
      mt = fmaxf(mt, __shfl_xor(mt, 4));
      mt = fmaxf(mt, __shfl_xor(mt, 8));
      const float mnew  = fmaxf(mrow[r], mt);
      const float scale = __expf(mrow[r] - mnew);
      mrow[r] = mnew;
      const float p0 = __expf(c0[r] - mnew);
      const float p1 = __expf(c1[r] - mnew);
      float ps = p0 + p1;
      ps += __shfl_xor(ps, 1);
      ps += __shfl_xor(ps, 2);
      ps += __shfl_xor(ps, 4);
      ps += __shfl_xor(ps, 8);
      lsum[r] = lsum[r] * scale + ps;
#pragma unroll
      for (int t = 0; t < 8; ++t) acc[t][r] *= scale;
      Pl[wave][(lhi * 8 + r) * LDSS + lr]      = (_Float16)p0;
      Pl[wave][(lhi * 8 + r) * LDSS + 16 + lr] = (_Float16)p1;
    }
    asm volatile("s_wait_dscnt 0" ::: "memory");

    Frag16 pf;
    pf.h[0] = *(const v8h*)(&Pl[wave][lr * LDSS + lhi * 16]);
    pf.h[1] = *(const v8h*)(&Pl[wave][lr * LDSS + lhi * 16 + 8]);

#pragma unroll
    for (int t = 0; t < 8; ++t) {
      Frag16 vf;
      const _Float16* vp = VT + (size_t)(kvh * 128 + t * 16 + lr) * 2048 + jb + lhi * 16;
      vf.f = *(const v16h*)vp;
      acc[t] = __builtin_amdgcn_wmma_f32_16x16x32_f16(false, pf.f, false, vf.f,
                                                      (short)0, acc[t], false, false);
    }
  }

#pragma unroll
  for (int t = 0; t < 8; ++t) {
#pragma unroll
    for (int r = 0; r < 8; ++r) {
      const int m = q0 + lhi * 8 + r;
      Out[(size_t)m * 4096 + head * 128 + t * 16 + lr] = (_Float16)(acc[t][r] / lsum[r]);
    }
  }
}

// ---------------------------------------------------------------------------
// Host-side orchestration
// ---------------------------------------------------------------------------
extern "C" void kernel_launch(void* const* d_in, const int* in_sizes, int n_in,
                              void* d_out, int out_size, void* d_ws, size_t ws_size,
                              hipStream_t stream) {
  (void)in_sizes; (void)n_in; (void)out_size; (void)ws_size;
  const float* x  = (const float*)d_in[0];
  const float* wq = (const float*)d_in[1];
  const float* wk = (const float*)d_in[2];
  const float* wv = (const float*)d_in[3];
  const float* wo = (const float*)d_in[4];
  const float* fc = (const float*)d_in[7];
  const float* fs = (const float*)d_in[8];
  float* out = (float*)d_out;

  char* ws = (char*)d_ws;
  _Float16* xh    = (_Float16*)(ws + 0);            // 2048*4096 f16 = 16 MB
  _Float16* wqkvh = (_Float16*)(ws + 16777216);     // 6144*4096 f16 = 48 MB
  _Float16* woh   = (_Float16*)(ws + 67108864);     // 4096*4096 f16 = 32 MB
  float*    qkvf  = (float*)(ws + 100663296);       // 2048*6144 f32 = 48 MB
  _Float16* qh    = (_Float16*)(ws + 150994944);    // 2048*4096 f16 = 16 MB
  _Float16* kh    = (_Float16*)(ws + 167772160);    // 2048*1024 f16 =  4 MB
  _Float16* vT    = (_Float16*)(ws + 171966464);    // 1024*2048 f16 =  4 MB
  _Float16* attnh = (_Float16*)(ws + 176160768);    // 2048*4096 f16 = 16 MB

  // fp32 -> f16 conversions (x; wq|wk|wv stacked into one [6144,4096]; wo)
  cvt_f16_kernel<<< (2048 * 4096 / 4) / 256, 256, 0, stream>>>(x,  xh);
  cvt_f16_kernel<<< (4096 * 4096 / 4) / 256, 256, 0, stream>>>(wq, wqkvh);
  cvt_f16_kernel<<< (1024 * 4096 / 4) / 256, 256, 0, stream>>>(wk, wqkvh + 16777216);
  cvt_f16_kernel<<< (1024 * 4096 / 4) / 256, 256, 0, stream>>>(wv, wqkvh + 20971520);
  cvt_f16_kernel<<< (4096 * 4096 / 4) / 256, 256, 0, stream>>>(wo, woh);

  // fused QKV projection: [2048,4096] x [6144,4096]^T -> [2048,6144]
  gemm_wmma_f16<<<dim3(6144 / GBN, 2048 / GBM), 256, 0, stream>>>(xh, wqkvh, qkvf,
                                                                  2048, 6144, 4096);
  // RoPE + f16 (+ softmax scale into Q), V transpose
  rope_q_kernel<<<(2048 * 2048) / 256, 256, 0, stream>>>(qkvf, fc, fs, qh);
  rope_k_kernel<<<(2048 * 512) / 256, 256, 0, stream>>>(qkvf, fc, fs, kh);
  transpose_v_kernel<<<(2048 * 1024) / 256, 256, 0, stream>>>(qkvf, vT);
  // Flash attention (32 q-tiles of 64 rows x 32 heads)
  flash_attn_kernel<<<dim3(2048 / 64, 32), 128, 0, stream>>>(qh, kh, vT, attnh);
  // Output projection
  gemm_wmma_f16<<<dim3(4096 / GBN, 2048 / GBM), 256, 0, stream>>>(attnh, woh, out,
                                                                  2048, 4096, 4096);
}
